// VAERNN_55654186222195
// MI455X (gfx1250) — matrix operand order repaired and checked by reading, not compiled, not run on hardware
//
#include <hip/hip_runtime.h>
#include <cstddef>
#include <cstdint>

// Problem dims (fixed by the reference)
#define HDIM  512      // hidden
#define GDIM  2048     // 4*H (gates i,f,g,o)
#define BATCH 256
#define SEQT  200
#define LATD  128
#define DDIM  2

typedef __attribute__((ext_vector_type(8)))  _Float16 v8h;
typedef __attribute__((ext_vector_type(16))) _Float16 v16h;
typedef __attribute__((ext_vector_type(8)))  float    v8f;

__device__ __forceinline__ float sigm(float x) { return 1.0f / (1.0f + __expf(-x)); }

// ---- WMMA A-fragment loader (ISA 7.12.2, wave32) --------------------------
// A (16x32 f16, MxK): lane<16  -> row=lane,    K = kb+0..7,  kb+16..23
//                     lane>=16 -> row=lane-16, K = kb+8..15, kb+24..31
__device__ __forceinline__ v16h load_afrag(const _Float16* __restrict__ base,
                                           int row, int kb) {
  const v8h lo = *(const v8h*)(base + row * HDIM + kb);
  const v8h hi = *(const v8h*)(base + row * HDIM + kb + 16);
  return __builtin_shufflevector(lo, hi, 0,1,2,3,4,5,6,7,8,9,10,11,12,13,14,15);
}

// ---- One LSTM time-step, fully fused: WMMA GEMM + wave-local pointwise ----
// Wave (mt,nt) computes the four gate tiles {nt, nt+H, nt+2H, nt+3H}.
// All 8 waves of a block share nt; the 4-gate B panel (64 rows x 256 K) is
// staged in LDS by the whole block per K-phase (ds_load_b128 inner loop).
// All 8 A fragments of a phase are loaded into live registers up front so
// their global latency overlaps the fill + barrier.
// Grid: 64 blocks x 256 threads (8 waves, wave32) -> 512 waves = 16mt x 32nt.
#define KCH    256            // K per LDS phase
#define PITCH  (KCH + 8)      // row pitch in halves (pad -> bank stride 4)

template <bool LAYER1>
__global__ __launch_bounds__(256)
void lstm_step_kernel(const _Float16* __restrict__ hprev,   // [B,H] f16
                      const _Float16* __restrict__ Whh,     // [4H,H] f16
                      const _Float16* __restrict__ yin,     // LAYER1: [B,H] f16
                      const _Float16* __restrict__ Wih16,   // LAYER1: [4H,H] f16
                      const float*    __restrict__ xsrc,    // !LAYER1: [B,T,D] f32
                      int                          xt,      // !LAYER1: time col; <0 => zero input
                      const float*    __restrict__ WihS,    // !LAYER1: [4H,2] f32
                      const float*    __restrict__ bias,    // [4H] f32 (bih+bhh)
                      float*          __restrict__ c,       // [B,H] f32 in/out
                      _Float16*       __restrict__ hout) {  // [B,H] f16 out
  __shared__ _Float16 sB[64 * PITCH];                       // ~33.8 KB

  const int lane = threadIdx.x & 31;
  const int wv   = threadIdx.x >> 5;
  const int nt   = blockIdx.x & 31;                      // hidden tile (shared in block)
  const int mt   = ((blockIdx.x >> 5) << 3) + wv;        // batch tile 0..15

  const int colL = lane & 15;
  const int arow = mt * 16 + colL;
  const int akb  = (lane >> 4) << 3;                     // 0 or 8
  const int ncol = nt * 16 + colL;                       // hidden column 0..511
  const int bkb  = (lane >> 4) << 3;                     // 0 or 8

  auto ldsb = [&](int row, int kb) -> v16h {             // B frag: 32B from LDS
    const v8h lo = *(const v8h*)(&sB[row * PITCH + kb]);
    const v8h hi = *(const v8h*)(&sB[row * PITCH + kb + 16]);
    return __builtin_shufflevector(lo, hi, 0,1,2,3,4,5,6,7,8,9,10,11,12,13,14,15);
  };

  v8f acc_i = {}, acc_f = {}, acc_g = {}, acc_o = {};

  constexpr int NPH = LAYER1 ? 4 : 2;                    // K phases of 256

#pragma unroll
  for (int p = 0; p < NPH; ++p) {
    const bool src1 = LAYER1 && (p >= 2);                // compile-time per phase
    const _Float16* W  = src1 ? Wih16 : Whh;
    const _Float16* hs = src1 ? yin   : hprev;
    const int       kb0 = (p & 1) * KCH;                 // k-offset within source

    // ---- A fragments for the whole phase: 8 live v16h (16x b128, one clause)
    v16h aph[8];
#pragma unroll
    for (int ks = 0; ks < KCH; ks += 32)
      aph[ks >> 5] = load_afrag(hs, arow, kb0 + ks + akb);

    // ---- cooperative fill: 64 rows x 256 halves (8x b128 per thread) ----
#pragma unroll
    for (int i = 0; i < 8; ++i) {
      const int chunk = (int)threadIdx.x + (i << 8);     // 0..2047
      const int row   = chunk >> 5;                      // 0..63 (gate*16+col)
      const int kc    = (chunk & 31) << 3;               // 0..248 step 8
      const int grow  = ((row >> 4) * HDIM) + nt * 16 + (row & 15);
      *(v8h*)(&sB[row * PITCH + kc]) =
          *(const v8h*)(W + (size_t)grow * HDIM + kb0 + kc);
    }
    __syncthreads();

    // ---- compute: 8 k-steps, B from LDS, A already resident ----
#pragma unroll
    for (int ks = 0; ks < KCH; ks += 32) {
      const int  kb = ks + bkb;
      const v16h a  = aph[ks >> 5];
      const v16h bi = ldsb(      colL, kb);
      const v16h bf = ldsb(16 + colL, kb);
      const v16h bg = ldsb(32 + colL, kb);
      const v16h bo = ldsb(48 + colL, kb);
      acc_i = __builtin_amdgcn_wmma_f32_16x16x32_f16(false, a, false, bi, (short)0, acc_i, false, false);
      acc_f = __builtin_amdgcn_wmma_f32_16x16x32_f16(false, a, false, bf, (short)0, acc_f, false, false);
      acc_g = __builtin_amdgcn_wmma_f32_16x16x32_f16(false, a, false, bg, (short)0, acc_g, false, false);
      acc_o = __builtin_amdgcn_wmma_f32_16x16x32_f16(false, a, false, bo, (short)0, acc_o, false, false);
    }
    __syncthreads();
  }

  const float b_i = bias[ncol];
  const float b_f = bias[ncol +   HDIM];
  const float b_g = bias[ncol + 2*HDIM];
  const float b_o = bias[ncol + 3*HDIM];

  float wi0 = 0.f, wi1 = 0.f, wf0 = 0.f, wf1 = 0.f;
  float wg0 = 0.f, wg1 = 0.f, wo0 = 0.f, wo1 = 0.f;
  if (!LAYER1) {  // rank-2 input projection folded into the epilogue
    wi0 = WihS[(ncol         ) * DDIM + 0]; wi1 = WihS[(ncol         ) * DDIM + 1];
    wf0 = WihS[(ncol +   HDIM) * DDIM + 0]; wf1 = WihS[(ncol +   HDIM) * DDIM + 1];
    wg0 = WihS[(ncol + 2*HDIM) * DDIM + 0]; wg1 = WihS[(ncol + 2*HDIM) * DDIM + 1];
    wo0 = WihS[(ncol + 3*HDIM) * DDIM + 0]; wo1 = WihS[(ncol + 3*HDIM) * DDIM + 1];
  }

  // C/D layout: VGPR r, lane -> (M = r + 8*(lane>=16), N = lane&15)
  const int mofs = (lane >= 16) ? 8 : 0;
#pragma unroll
  for (int r = 0; r < 8; ++r) {
    const int brow = mt * 16 + r + mofs;
    float gi = acc_i[r] + b_i;
    float gf = acc_f[r] + b_f;
    float gg = acc_g[r] + b_g;
    float go = acc_o[r] + b_o;
    if (!LAYER1 && xt >= 0) {
      const float x0 = xsrc[((size_t)brow * SEQT + xt) * DDIM + 0];
      const float x1 = xsrc[((size_t)brow * SEQT + xt) * DDIM + 1];
      gi += x0 * wi0 + x1 * wi1;  gf += x0 * wf0 + x1 * wf1;
      gg += x0 * wg0 + x1 * wg1;  go += x0 * wo0 + x1 * wo1;
    }
    const size_t idx = (size_t)brow * HDIM + ncol;
    const float cv = c[idx];
    const float iv = sigm(gi), fv = sigm(gf), ov = sigm(go);
    const float gv = tanhf(gg);
    const float cn = fv * cv + iv * gv;
    const float hv = ov * tanhf(cn);
    c[idx]    = cn;
    hout[idx] = (_Float16)hv;
  }
}

// ---- latent heads: mu, logvar, z ------------------------------------------
__global__ __launch_bounds__(256)
void heads_kernel(const _Float16* __restrict__ hT, const float* __restrict__ muW,
                  const float* __restrict__ mub, const float* __restrict__ lvW,
                  const float* __restrict__ lvb, const float* __restrict__ eps,
                  float* __restrict__ out_mu, float* __restrict__ out_lv,
                  float* __restrict__ out_z, float* __restrict__ z_ws) {
  const int idx = blockIdx.x * blockDim.x + threadIdx.x;
  if (idx >= BATCH * LATD) return;
  const int b = idx / LATD, j = idx % LATD;
  float smu = mub[j], slv = lvb[j];
  for (int k = 0; k < HDIM; ++k) {
    const float hv = (float)hT[(size_t)b * HDIM + k];
    smu += hv * muW[(size_t)j * HDIM + k];
    slv += hv * lvW[(size_t)j * HDIM + k];
  }
  const float zv = smu + eps[idx] * __expf(0.5f * slv);
  out_mu[idx] = smu;  out_lv[idx] = slv;  out_z[idx] = zv;  z_ws[idx] = zv;
}

// ---- decoder init states from z -------------------------------------------
__global__ __launch_bounds__(256)
void dec_init_kernel(const float* __restrict__ z, const float* __restrict__ l2hW,
                     const float* __restrict__ l2hb, const float* __restrict__ l2cW,
                     const float* __restrict__ l2cb,
                     _Float16* __restrict__ h16_0, _Float16* __restrict__ h16_1,
                     float* __restrict__ c0_, float* __restrict__ c1_) {
  const int idx = blockIdx.x * blockDim.x + threadIdx.x;
  if (idx >= BATCH * 2 * HDIM) return;
  const int b = idx / (2 * HDIM), lh = idx % (2 * HDIM);
  float sh = l2hb[lh], sc = l2cb[lh];
  for (int k = 0; k < LATD; ++k) {
    const float zv = z[(size_t)b * LATD + k];
    sh += zv * l2hW[(size_t)lh * LATD + k];
    sc += zv * l2cW[(size_t)lh * LATD + k];
  }
  const int l = lh / HDIM, hc = lh % HDIM;
  const size_t o = (size_t)b * HDIM + hc;
  if (l == 0) { h16_0[o] = (_Float16)sh; c0_[o] = sc; }
  else        { h16_1[o] = (_Float16)sh; c1_[o] = sc; }
}

// ---- readout: recon[b,t,d] = d1[t,b,:] . out_W[d,:] + out_b[d] ------------
__global__ __launch_bounds__(256)
void recon_kernel(const _Float16* __restrict__ d1seq, const float* __restrict__ outW,
                  const float* __restrict__ outb, float* __restrict__ out) {
  const int idx = blockIdx.x * blockDim.x + threadIdx.x;
  if (idx >= BATCH * SEQT * DDIM) return;
  const int d = idx % DDIM, bt = idx / DDIM, t = bt % SEQT, b = bt / SEQT;
  float s = outb[d];
  const _Float16* hrow = d1seq + ((size_t)t * BATCH + b) * HDIM;
  for (int k = 0; k < HDIM; ++k) s += (float)hrow[k] * outW[(size_t)d * HDIM + k];
  out[idx] = s;
}

// ---- weight prep -----------------------------------------------------------
__global__ __launch_bounds__(256)
void cvt_f16_kernel(const float* __restrict__ src, _Float16* __restrict__ dst, int n) {
  const int i = blockIdx.x * blockDim.x + threadIdx.x;
  if (i < n) dst[i] = (_Float16)src[i];
}
__global__ __launch_bounds__(256)
void bias_sum_kernel(const float* __restrict__ a, const float* __restrict__ b,
                     float* __restrict__ dst, int n) {
  const int i = blockIdx.x * blockDim.x + threadIdx.x;
  if (i < n) dst[i] = a[i] + b[i];
}

// ---------------------------------------------------------------------------
extern "C" void kernel_launch(void* const* d_in, const int* in_sizes, int n_in,
                              void* d_out, int out_size, void* d_ws, size_t ws_size,
                              hipStream_t stream) {
  (void)in_sizes; (void)n_in; (void)out_size; (void)ws_size;
  const float* x       = (const float*)d_in[0];
  const float* eps     = (const float*)d_in[1];
  const float* W_ih[4] = {(const float*)d_in[2],  (const float*)d_in[6],
                          (const float*)d_in[10], (const float*)d_in[14]};
  const float* W_hh[4] = {(const float*)d_in[3],  (const float*)d_in[7],
                          (const float*)d_in[11], (const float*)d_in[15]};
  const float* b_ih[4] = {(const float*)d_in[4],  (const float*)d_in[8],
                          (const float*)d_in[12], (const float*)d_in[16]};
  const float* b_hh[4] = {(const float*)d_in[5],  (const float*)d_in[9],
                          (const float*)d_in[13], (const float*)d_in[17]};
  const float* l2hW = (const float*)d_in[18]; const float* l2hb = (const float*)d_in[19];
  const float* l2cW = (const float*)d_in[20]; const float* l2cb = (const float*)d_in[21];
  const float* muW  = (const float*)d_in[22]; const float* mub  = (const float*)d_in[23];
  const float* lvW  = (const float*)d_in[24]; const float* lvb  = (const float*)d_in[25];
  const float* outW = (const float*)d_in[26]; const float* outb = (const float*)d_in[27];

  // ---- workspace carve-up (256B aligned) ----
  char* ws = (char*)d_ws;
  size_t off = 0;
  auto carve = [&](size_t bytes) -> char* {
    char* p = ws + off;
    off = (off + bytes + 255) & ~(size_t)255;
    return p;
  };
  const size_t BH   = (size_t)BATCH * HDIM;
  const size_t SEQB = (size_t)SEQT * BH;
  _Float16* enc0_seq = (_Float16*)carve(SEQB * 2);
  _Float16* dec0_seq = (_Float16*)carve(SEQB * 2);
  _Float16* dec1_seq = (_Float16*)carve(SEQB * 2);
  _Float16* e1h[2]   = {(_Float16*)carve(BH * 2), (_Float16*)carve(BH * 2)};
  _Float16* zero_h   = (_Float16*)carve(BH * 2);
  _Float16* dih[2]   = {(_Float16*)carve(BH * 2), (_Float16*)carve(BH * 2)};
  float* c_enc0 = (float*)carve(BH * 4);
  float* c_enc1 = (float*)carve(BH * 4);
  float* c_dec0 = (float*)carve(BH * 4);
  float* c_dec1 = (float*)carve(BH * 4);
  float* z_ws   = (float*)carve((size_t)BATCH * LATD * 4);
  const size_t WN = (size_t)GDIM * HDIM;
  _Float16* Whh16[4]; for (int l = 0; l < 4; ++l) Whh16[l] = (_Float16*)carve(WN * 2);
  _Float16* Wih16_e1 = (_Float16*)carve(WN * 2);     // enc1 Wih (H input)
  _Float16* Wih16_d1 = (_Float16*)carve(WN * 2);     // dec1 Wih (H input)
  float* bias[4]; for (int l = 0; l < 4; ++l) bias[l] = (float*)carve(GDIM * 4);

  // ---- output layout: recon | mu | logvar | z ----
  float* out_recon = (float*)d_out;
  float* out_mu    = out_recon + (size_t)BATCH * SEQT * DDIM;
  float* out_lv    = out_mu + (size_t)BATCH * LATD;
  float* out_z     = out_lv + (size_t)BATCH * LATD;

  // ---- prep: zero states, f16 weights, fused biases ----
  hipMemsetAsync(zero_h, 0, BH * 2, stream);
  hipMemsetAsync(e1h[0], 0, BH * 2, stream);
  hipMemsetAsync(c_enc0, 0, BH * 4, stream);
  hipMemsetAsync(c_enc1, 0, BH * 4, stream);
  const int cvtGrid = (int)(WN / 256);
  for (int l = 0; l < 4; ++l)
    cvt_f16_kernel<<<cvtGrid, 256, 0, stream>>>(W_hh[l], Whh16[l], (int)WN);
  cvt_f16_kernel<<<cvtGrid, 256, 0, stream>>>(W_ih[1], Wih16_e1, (int)WN);
  cvt_f16_kernel<<<cvtGrid, 256, 0, stream>>>(W_ih[3], Wih16_d1, (int)WN);
  for (int l = 0; l < 4; ++l)
    bias_sum_kernel<<<GDIM / 256, 256, 0, stream>>>(b_ih[l], b_hh[l], bias[l], GDIM);

  // ---- encoder layer 0 (x input, D=2 folded into epilogue) ----
  for (int t = 0; t < SEQT; ++t) {
    const _Float16* hp = (t == 0) ? zero_h : enc0_seq + (size_t)(t - 1) * BH;
    lstm_step_kernel<false><<<64, 256, 0, stream>>>(
        hp, Whh16[0], nullptr, nullptr, x, t, W_ih[0], bias[0],
        c_enc0, enc0_seq + (size_t)t * BH);
  }
  // ---- encoder layer 1 (fused y@Wih^T; h double-buffered) ----
  for (int t = 0; t < SEQT; ++t) {
    lstm_step_kernel<true><<<64, 256, 0, stream>>>(
        e1h[t & 1], Whh16[1], enc0_seq + (size_t)t * BH, Wih16_e1,
        nullptr, -1, nullptr, bias[1], c_enc1, e1h[(t + 1) & 1]);
  }
  const _Float16* hT1 = e1h[SEQT & 1];

  // ---- heads + reparameterization ----
  heads_kernel<<<(BATCH * LATD + 255) / 256, 256, 0, stream>>>(
      hT1, muW, mub, lvW, lvb, eps, out_mu, out_lv, out_z, z_ws);

  // ---- decoder init states ----
  dec_init_kernel<<<(BATCH * 2 * HDIM + 255) / 256, 256, 0, stream>>>(
      z_ws, l2hW, l2hb, l2cW, l2cb, dih[0], dih[1], c_dec0, c_dec1);

  // ---- decoder layer 0 (teacher forcing: input at t is x[:, t-1], t=0 -> 0) ----
  for (int t = 0; t < SEQT; ++t) {
    const _Float16* hp = (t == 0) ? dih[0] : dec0_seq + (size_t)(t - 1) * BH;
    lstm_step_kernel<false><<<64, 256, 0, stream>>>(
        hp, Whh16[2], nullptr, nullptr, x, t - 1, W_ih[2], bias[2],
        c_dec0, dec0_seq + (size_t)t * BH);
  }
  // ---- decoder layer 1 ----
  for (int t = 0; t < SEQT; ++t) {
    const _Float16* hp = (t == 0) ? dih[1] : dec1_seq + (size_t)(t - 1) * BH;
    lstm_step_kernel<true><<<64, 256, 0, stream>>>(
        hp, Whh16[3], dec0_seq + (size_t)t * BH, Wih16_d1,
        nullptr, -1, nullptr, bias[3], c_dec1, dec1_seq + (size_t)t * BH);
  }

  // ---- readout ----
  recon_kernel<<<(BATCH * SEQT * DDIM + 255) / 256, 256, 0, stream>>>(
      dec1_seq, outW, outb, out_recon);
}
